// my_lstm_1322849927328
// MI455X (gfx1250) — compile-verified, matrix operand order checked by reading, not run on hardware
//
#include <hip/hip_runtime.h>
#include <hip/hip_bf16.h>
#include <math.h>

typedef _Float16 f16;
typedef __attribute__((ext_vector_type(16))) _Float16 v16h;
typedef __attribute__((ext_vector_type(4)))  _Float16 v4h;
typedef __attribute__((ext_vector_type(8)))  float    v8f;

typedef unsigned int u32x4 __attribute__((ext_vector_type(4)));
typedef int          i32x8 __attribute__((ext_vector_type(8)));
typedef int          i32x4 __attribute__((ext_vector_type(4)));

constexpr int kB   = 64;    // batch
constexpr int kT   = 256;   // timesteps
constexpr int kD   = 3;     // input dim
constexpr int kH   = 512;   // hidden
constexpr int kNWG = 32;    // persistent workgroups
constexpr int kNH  = 16;    // hidden cols per WG
constexpr int kGR  = 64;    // gate rows per WG (4 * kNH)
constexpr int kLDH = 520;   // padded LDS stride for f16 rows (halves) = 512 + 8
constexpr int kLDG = 68;    // padded gates stride (floats)
constexpr int kKC  = kH / 32;  // 16 K-chunks of 32
constexpr int kTHREADS = 256;

// Load one 16x32 f16 WMMA operand fragment from LDS (row-major, stride ld).
// CDNA5 ISA 16-bit A 16x32 layout: lanes 0-15 hold row M=lane,
// K = {0..7, 16..23}; lanes 16-31 hold row M=lane-16, K = {8..15, 24..31}.
__device__ inline v16h frag_ld(const f16* __restrict__ p, int ld) {
    const int lane = threadIdx.x & 31;
    const f16* r = p + (lane & 15) * ld + ((lane >> 4) << 3);
    v16h out;
#pragma unroll
    for (int i = 0; i < 8; ++i) {
        out[i]     = r[i];
        out[i + 8] = r[i + 16];
    }
    return out;
}

__global__ void lstm_init_kernel(f16* __restrict__ hbuf, unsigned* __restrict__ bar) {
    int tid = blockIdx.x * blockDim.x + threadIdx.x;
    if (tid < 2) bar[tid] = 0u;
    for (int i = tid; i < 2 * kB * kH; i += blockDim.x * gridDim.x)
        hbuf[i] = (f16)0.f;
}

__global__ __launch_bounds__(kTHREADS, 1)
void lstm_persistent_kernel(const float* __restrict__ x,
                            const float* __restrict__ fx,
                            const float* __restrict__ x2h_w,
                            const float* __restrict__ x2h_b,
                            const float* __restrict__ h2h_w,
                            const float* __restrict__ h2h_b,
                            const float* __restrict__ fc_w,
                            const float* __restrict__ fc_b,
                            float* __restrict__ out,
                            f16* __restrict__ hbuf,      // [2][kB][kH] ping-pong
                            unsigned* __restrict__ bar)  // {count, generation}
{
    __shared__ f16   w_lds[kGR * kLDH];     // 65 KB: this WG's 64x512 f16 weight slab
    __shared__ f16   h_lds[kB * kLDH];      // 65 KB: staged h_t (64x512 f16, padded rows)
    __shared__ float gates_lds[kB * kLDG];  // 17 KB: pre-activation gates [64][64]
    __shared__ float c_lds[kB * kNH];       // 4 KB : fp32 cell state slice
    __shared__ float bias_lds[kGR];         // x2h_b + h2h_b slice
    __shared__ float xw_lds[kGR * kD];      // x2h_w slice (D=3)

    const int tid  = threadIdx.x;
    const int wg   = blockIdx.x;
    const int lane = tid & 31;
    const int wave = tid >> 5;

    // Wave -> output tiling: wave owns N-tile ni (16 gate rows == gate index
    // ni of this WG's hidden slice) and two M-tiles (16 batch rows each).
    const int ni  = wave & 3;
    const int mi0 = (wave >> 2) << 1;

    // ---- One-time setup: convert + stage this WG's weight slab into LDS ----
    // Slab row r = g*16 + hc  <->  global h2h_w row G = g*512 + wg*16 + hc.
    // Weights stay LDS-resident for all 256 steps (register-hoisting them
    // spills to scratch: 128 VGPRs of B fragments exceed the VGPR budget).
    for (int i = tid; i < kGR * kH; i += kTHREADS) {
        int r = i >> 9;            // / 512
        int c = i & (kH - 1);
        int G = (r >> 4) * kH + wg * kNH + (r & 15);
        w_lds[r * kLDH + c] = (f16)h2h_w[G * kH + c];
    }
    for (int r = tid; r < kGR; r += kTHREADS) {
        int G = (r >> 4) * kH + wg * kNH + (r & 15);
        bias_lds[r] = x2h_b[G] + h2h_b[G];
        xw_lds[r * kD + 0] = x2h_w[G * kD + 0];
        xw_lds[r * kD + 1] = x2h_w[G * kD + 1];
        xw_lds[r * kD + 2] = x2h_w[G * kD + 2];
    }
    for (int i = tid; i < kB * kNH; i += kTHREADS) c_lds[i] = 0.f;
    __syncthreads();

    // ---- Tensor Data Mover descriptors (CDNA5 ISA) ----
    // One TDM op stages the whole 64x512 f16 h tile global->LDS per step,
    // inserting 4 DWORDs of padding after every 256 DWORDs (one 512-half row)
    // to produce the 520-half padded LDS stride.
    const unsigned lds_h_addr = (unsigned)(uintptr_t)(&h_lds[0]); // LDS byte offset
    const unsigned long long ga0 = (unsigned long long)(uintptr_t)hbuf;
    const unsigned long long ga1 = ga0 + (unsigned long long)kB * kH * sizeof(f16);

    u32x4 g0e, g0o;
    {
        g0e[0] = 1u;                                   // count=1, no gather
        g0e[1] = lds_h_addr;                           // lds_addr [63:32]
        g0e[2] = (unsigned)(ga0 & 0xFFFFFFFFull);      // global_addr [95:64]
        g0e[3] = (unsigned)((ga0 >> 32) & 0x1FFFFFFull) | (2u << 30); // addr hi | type=2
        g0o[0] = 1u;
        g0o[1] = lds_h_addr;
        g0o[2] = (unsigned)(ga1 & 0xFFFFFFFFull);
        g0o[3] = (unsigned)((ga1 >> 32) & 0x1FFFFFFull) | (2u << 30);
    }
    i32x8 g1;
    {
        // [15:0] wg_mask=0 | [17:16] data_size=1 (2B) | [20] pad_enable
        // [24:22] pad_interval=7 (256 DW) | [31:25] pad_amount=3 (4 DW)
        g1[0] = (1 << 16) | (1 << 20) | (7 << 22) | (3 << 25);
        g1[1] = (kH & 0xFFFF) << 16;                   // tensor_dim0 lo16 @ [63:48]
        g1[2] = (kH >> 16) | ((kB & 0xFFFF) << 16);    // tensor_dim0 hi | tensor_dim1 lo
        g1[3] = (kB >> 16) | (kH << 16);               // tensor_dim1 hi | tile_dim0=512
        g1[4] = kB;                                    // tile_dim1=64, tile_dim2=0
        g1[5] = kH;                                    // tensor_dim0_stride lo32 = 512
        g1[6] = 0;
        g1[7] = 0;
    }
    i32x4 g2 = {0, 0, 0, 0};
    i32x4 g3 = {0, 0, 0, 0};
    i32x8 g4 = {0, 0, 0, 0, 0, 0, 0, 0};   // trailing group (6-arg builtin form)

    for (int t = 0; t < kT; ++t) {
        f16* hw = hbuf + (size_t)((t & 1) ^ 1) * kB * kH;  // write h_{t+1}

        // ---- Phase 1: TDM-stage full h tile into LDS (one DMA per WG) ----
        if (tid < 32) {
            u32x4 g0 = (t & 1) ? g0o : g0e;
            __builtin_amdgcn_tensor_load_to_lds(g0, g1, g2, g3, g4, 0);
            __builtin_amdgcn_s_wait_tensorcnt(0);
        }
        __syncthreads();

        // ---- Phase 2: gates_pre = h @ W_slab^T  via v_wmma_f32_16x16x32_f16 ----
        v8f acc0 = {};
        v8f acc1 = {};
#pragma unroll 4
        for (int k = 0; k < kKC; ++k) {
            v16h bfrag = frag_ld(w_lds + (ni << 4) * kLDH + (k << 5), kLDH);
            v16h a0    = frag_ld(h_lds + (mi0 << 4) * kLDH + (k << 5), kLDH);
            acc0 = __builtin_amdgcn_wmma_f32_16x16x32_f16(
                false, a0, false, bfrag, (short)0, acc0, false, false);
            v16h a1    = frag_ld(h_lds + ((mi0 + 1) << 4) * kLDH + (k << 5), kLDH);
            acc1 = __builtin_amdgcn_wmma_f32_16x16x32_f16(
                false, a1, false, bfrag, (short)0, acc1, false, false);
        }
        // Scatter fragments (ISA C/D layout: lane n = N, VGPR v = M row) to LDS.
        {
            int nbase = (ni << 4) + (lane & 15);
            int mofs  = (lane >> 4) << 3;
#pragma unroll
            for (int v = 0; v < 8; ++v) {
                gates_lds[((mi0 << 4)       + mofs + v) * kLDG + nbase] = acc0[v];
                gates_lds[(((mi0 + 1) << 4) + mofs + v) * kLDG + nbase] = acc1[v];
            }
        }
        __syncthreads();

        // ---- Phase 3: fold in x-projection + bias, activations, cell update ----
        {
            int b     = tid & 63;
            int hc_hi = tid >> 6;  // 0..3 -> hidden cols hc_hi*4 .. hc_hi*4+3
            const float* xr = x + ((size_t)b * kT + t) * kD;
            float x0 = xr[0], x1 = xr[1], x2 = xr[2];
            v4h hv;
#pragma unroll
            for (int q = 0; q < 4; ++q) {
                int hc = (hc_hi << 2) + q;
                float pre[4];
#pragma unroll
                for (int g = 0; g < 4; ++g) {
                    int r = (g << 4) + hc;
                    pre[g] = gates_lds[b * kLDG + r] + bias_lds[r]
                           + x0 * xw_lds[r * kD + 0]
                           + x1 * xw_lds[r * kD + 1]
                           + x2 * xw_lds[r * kD + 2];
                }
                float ig = 1.f / (1.f + __expf(-pre[0]));
                float fg = 1.f / (1.f + __expf(-pre[1]));
                float gg = tanhf(pre[2]);
                float og = 1.f / (1.f + __expf(-pre[3]));
                float cn = c_lds[b * kNH + hc] * fg + ig * gg;
                c_lds[b * kNH + hc] = cn;
                hv[q] = (f16)(og * tanhf(cn));
            }
            // One 8B store for the 4 consecutive f16 hidden values.
            *(v4h*)(hw + (size_t)b * kH + wg * kNH + (hc_hi << 2)) = hv;
        }

        // ---- Phase 4: grid-wide sense barrier (persistent-kernel sync) ----
        __syncthreads();
        if (tid == 0) {
            __threadfence();
            unsigned old = atomicAdd(&bar[0], 1u);
            if (old == (unsigned)(gridDim.x - 1)) {
                bar[0] = 0u;
                __threadfence();
                atomicAdd(&bar[1], 1u);
            } else {
                while (__hip_atomic_load(&bar[1], __ATOMIC_RELAXED,
                                         __HIP_MEMORY_SCOPE_AGENT) < (unsigned)(t + 1)) {
                    __builtin_amdgcn_s_sleep(1);
                }
                __threadfence();
            }
        }
        __syncthreads();
    }

    // ---- FC head: out[b] = h_T[b,:] . fc_w[0:512] + fx[b]*fc_w[512] + fc_b ----
    // T=256 is even -> final h lives in buffer 0.
    if (wg == 0) {
        const f16* hf = hbuf;
        for (int b = tid; b < kB; b += kTHREADS) {
            const f16* row = hf + b * kH;
            float s = 0.f;
#pragma unroll 8
            for (int j = 0; j < kH; ++j) s += (float)row[j] * fc_w[j];
            s += fx[b] * fc_w[kH];
            s += fc_b[0];
            out[b] = s;
        }
    }
}

extern "C" void kernel_launch(void* const* d_in, const int* in_sizes, int n_in,
                              void* d_out, int out_size, void* d_ws, size_t ws_size,
                              hipStream_t stream) {
    const float* x     = (const float*)d_in[0];
    const float* fx    = (const float*)d_in[1];
    const float* x2h_w = (const float*)d_in[2];
    const float* x2h_b = (const float*)d_in[3];
    const float* h2h_w = (const float*)d_in[4];
    const float* h2h_b = (const float*)d_in[5];
    const float* fc_w  = (const float*)d_in[6];
    const float* fc_b  = (const float*)d_in[7];

    unsigned char* ws = (unsigned char*)d_ws;
    unsigned* bar = (unsigned*)ws;               // 2 counters (256B reserved)
    f16* hbuf     = (f16*)(ws + 256);            // 2 * 64 * 512 f16 ping-pong

    lstm_init_kernel<<<64, 256, 0, stream>>>(hbuf, bar);
    lstm_persistent_kernel<<<kNWG, kTHREADS, 0, stream>>>(
        x, fx, x2h_w, x2h_b, h2h_w, h2h_b, fc_w, fc_b,
        (float*)d_out, hbuf, bar);
}
